// SS2DKANBlock_16166256902891
// MI455X (gfx1250) — matrix operand order repaired and checked by reading, not compiled, not run on hardware
//
#include <hip/hip_runtime.h>
#include <hip/hip_bf16.h>
#include <math.h>

// ---------------- problem constants (from setup_inputs) ----------------
constexpr int BB = 4;          // batch
constexpr int HH = 64, WW = 64;
constexpr int CC = 96;         // channels
constexpr int LLEN = HH * WW;  // 4096
constexpr int MROWS = BB * LLEN;   // 16384
constexpr int DI = 2 * CC;     // 192
constexpr int KDIR = 4;        // scan directions
constexpr int NST = 16;        // state dim
constexpr int RR = 6;          // dt rank
constexpr int CPROJ = RR + 2 * NST;  // 38
constexpr int NB = 8;          // KAN spline bases
constexpr int KSPL = CC * NB;  // 768

typedef __attribute__((ext_vector_type(16))) _Float16 v16h;
typedef __attribute__((ext_vector_type(8)))  _Float16 v8h;
typedef __attribute__((ext_vector_type(4)))  _Float16 v4h;
typedef __attribute__((ext_vector_type(8)))  float    v8f;

__device__ __forceinline__ float silu_f(float v) { return v / (1.0f + __expf(-v)); }
__device__ __forceinline__ float4 ld4(const float* p) { return *(const float4*)p; }
__device__ __forceinline__ v8h ld8h(const _Float16* p) { return *(const v8h*)p; }
__device__ __forceinline__ float getc(const float4& v, int s) {
  return s == 0 ? v.x : s == 1 ? v.y : s == 2 ? v.z : v.w;
}
__device__ __forceinline__ void setc(float4& v, int s, float x) {
  if (s == 0) v.x = x; else if (s == 1) v.y = x; else if (s == 2) v.z = x; else v.w = x;
}

// ---------------- f32 -> f16 conversion (weights; 4-wide) ----------------
__global__ void k_cvt_f16(const float* __restrict__ in, _Float16* __restrict__ out, int n4) {
  int i = blockIdx.x * blockDim.x + threadIdx.x;
  if (i >= n4) return;
  float4 v = ld4(in + (size_t)i * 4);
  v4h o;
  o[0] = (_Float16)v.x; o[1] = (_Float16)v.y; o[2] = (_Float16)v.z; o[3] = (_Float16)v.w;
  *(v4h*)(out + (size_t)i * 4) = o;
}

// ---------------- LayerNorm: one wave per row; optional f32 / f16 outputs ----------------
__global__ void k_layernorm(const float* __restrict__ in, const float* __restrict__ g,
                            const float* __restrict__ b, float* __restrict__ outF,
                            _Float16* __restrict__ outH, int rows, int C) {
  int wv = (blockIdx.x * blockDim.x + threadIdx.x) >> 5;
  int lane = threadIdx.x & 31;
  if (wv >= rows) return;
  const float* x = in + (size_t)wv * C;
  float s = 0.f, s2 = 0.f;
  for (int c = lane; c < C; c += 32) { float v = x[c]; s += v; s2 += v * v; }
  for (int o = 16; o; o >>= 1) { s += __shfl_xor(s, o, 32); s2 += __shfl_xor(s2, o, 32); }
  float mu = s / C;
  float rs = rsqrtf(s2 / C - mu * mu + 1e-5f);
  for (int c = lane; c < C; c += 32) {
    float r = (x[c] - mu) * rs * g[c] + b[c];
    if (outF) outF[(size_t)wv * C + c] = r;
    if (outH) outH[(size_t)wv * C + c] = (_Float16)r;
  }
}

// ---------------- WMMA GEMM: C[M,N] = A[M,K] @ W[N,K]^T, f16 operands, f32 accum ----------------
// NT 16x16 N-tiles per wave (A fragment reused NT times; NT=6 covers a full
// 96-wide N group so each A row streams from HBM once). K templated -> fully
// unrolled; all b128 loads of a step issued before fragment use (MLP + pipelining).
// MODE: 0 = store, 1 = store + resid, 2 = accumulate into C.
// Tile guard is wave-uniform so EXEC is all-ones at every WMMA.
template <int NT, int MODE, int KK>
__global__ void k_wmma_gemm(const _Float16* __restrict__ A, const _Float16* __restrict__ Wt,
                            const float* __restrict__ resid, float* __restrict__ C,
                            int M, int N) {
  const int lane = threadIdx.x & 31;
  const int wave = threadIdx.x >> 5;
  const int groupsN = N / (16 * NT);
  const int total = (M >> 4) * groupsN;
  int grp = blockIdx.x * (blockDim.x >> 5) + wave;
  if (grp >= total) return;
  const int tm = grp / groupsN;
  const int tn0 = (grp % groupsN) * NT;
  const int l15 = lane & 15;
  const int kh = lane >> 4;
  const _Float16* Arow = A + (size_t)((tm << 4) + l15) * KK;
  const _Float16* Wr[NT];
#pragma unroll
  for (int t = 0; t < NT; ++t) Wr[t] = Wt + (size_t)(((tn0 + t) << 4) + l15) * KK;
  v8f acc[NT];
#pragma unroll
  for (int t = 0; t < NT; ++t) acc[t] = (v8f){};

  constexpr int STEPS = KK / 32;
#pragma unroll
  for (int s = 0; s < STEPS; ++s) {
    const int k0 = s * 32;
    // Issue ALL loads of this step first (2 + 2*NT b128 loads, no cvt chains).
    // A 16x32 f16 fragment (ISA 7.12.2): per lane two contiguous 8-half runs.
    v8h a0 = ld8h(Arow + k0 + (kh << 3));
    v8h a1 = ld8h(Arow + k0 + 16 + (kh << 3));
    v8h blo[NT], bhi[NT];
#pragma unroll
    for (int t = 0; t < NT; ++t) {
      // B 32x16 f16 fragment: per lane two contiguous 8-half runs (same k order as A).
      blo[t] = ld8h(Wr[t] + k0 + (kh << 3));
      bhi[t] = ld8h(Wr[t] + k0 + 16 + (kh << 3));
    }
    if (s + 1 < STEPS) __builtin_prefetch(Arow + k0 + 32 + (kh << 3), 0, 3);
    v16h a;
#pragma unroll
    for (int e = 0; e < 8; ++e) { a[e] = a0[e]; a[e + 8] = a1[e]; }
#pragma unroll
    for (int t = 0; t < NT; ++t) {
      v16h b;
#pragma unroll
      for (int e = 0; e < 8; ++e) { b[e] = blo[t][e]; b[e + 8] = bhi[t][e]; }
      acc[t] = __builtin_amdgcn_wmma_f32_16x16x32_f16(false, a, false, b, (short)0,
                                                      acc[t], false, false);
    }
  }
#pragma unroll
  for (int t = 0; t < NT; ++t) {
    int col = ((tn0 + t) << 4) + l15;
#pragma unroll
    for (int r = 0; r < 8; ++r) {
      int row = (tm << 4) + (kh << 3) + r;   // C/D layout: m = r + 8*khalf
      size_t idx = (size_t)row * N + col;
      float v = acc[t][r];
      if constexpr (MODE == 1) v += resid[idx];
      if constexpr (MODE == 2) v += C[idx];
      C[idx] = v;
    }
  }
}

// ---------------- depthwise 3x3 conv (+bias) + SiLU, from xz[...,:Di] -> img(B,Di,H,W) ----------------
// d is the fastest thread index so the 9 xz reads are fully coalesced.
__global__ void k_dwconv_silu(const float* __restrict__ xz, const float* __restrict__ w,
                              const float* __restrict__ bias, float* __restrict__ img) {
  int idx = blockIdx.x * blockDim.x + threadIdx.x;
  if (idx >= BB * DI * HH * WW) return;
  int d = idx % DI;
  int rest = idx / DI;
  int j = rest & (WW - 1);
  int i = (rest >> 6) & (HH - 1);
  int b = rest >> 12;
  float acc = bias[d];
#pragma unroll
  for (int ki = 0; ki < 3; ++ki) {
#pragma unroll
    for (int kj = 0; kj < 3; ++kj) {
      int ii = i + ki - 1, jj = j + kj - 1;
      if (ii >= 0 && ii < HH && jj >= 0 && jj < WW)
        acc += w[d * 9 + ki * 3 + kj] *
               xz[((size_t)b * LLEN + ii * WW + jj) * (2 * DI) + d];
    }
  }
  img[((size_t)(b * DI + d) * HH + i) * WW + j] = silu_f(acc);
}

// ---------------- build xs(B,4,Di,L): diag / anti-diag gathers + flips ----------------
__global__ void k_build_xs(const float* __restrict__ img, float* __restrict__ xs) {
  int idx = blockIdx.x * blockDim.x + threadIdx.x;
  if (idx >= BB * DI * LLEN) return;
  int l = idx % LLEN;
  int d = (idx / LLEN) % DI;
  int b = idx / (DI * LLEN);
  int i = l & (HH - 1);       // l = j*H + i (column-major flatten of transposed gather)
  int j = l >> 6;
  const float* ip = img + ((size_t)(b * DI + d) * HH + i) * WW;
  float v0 = ip[(i + j) & (WW - 1)];            // diag gather
  float v1 = ip[(j - i + WW) & (WW - 1)];       // anti-diag gather
  size_t base = ((size_t)(b * KDIR) * DI + d) * LLEN;
  size_t str = (size_t)DI * LLEN;
  xs[base + 0 * str + l] = v0;
  xs[base + 1 * str + l] = v1;
  xs[base + 2 * str + (LLEN - 1 - l)] = v0;     // flip
  xs[base + 3 * str + (LLEN - 1 - l)] = v1;     // flip
}

// ---------------- x_proj einsum: xdbl[b,k,c,l] = sum_d xs[b,k,d,l]*w[k,c,d] (LDS-staged) ----------------
constexpr int XT = 64;  // l-tile
__global__ void k_xproj(const float* __restrict__ xs, const float* __restrict__ w,
                        float* __restrict__ xdbl) {
  __shared__ float sm[DI][XT + 1];
  int bk = blockIdx.y;             // b*4 + k
  int l0 = blockIdx.x * XT;
  int k = bk & 3;
  const float* xsp = xs + (size_t)bk * DI * LLEN;
  for (int t = threadIdx.x; t < DI * XT; t += blockDim.x) {
    int d = t / XT, ll = t % XT;
    sm[d][ll] = xsp[(size_t)d * LLEN + l0 + ll];
  }
  __syncthreads();
  const float* wk = w + (size_t)k * CPROJ * DI;
  float* outp = xdbl + (size_t)bk * CPROJ * LLEN;
  for (int t = threadIdx.x; t < CPROJ * XT; t += blockDim.x) {
    int c = t / XT, ll = t % XT;
    float acc = 0.f;
    const float* wr = wk + c * DI;
#pragma unroll 8
    for (int d = 0; d < DI; ++d) acc += sm[d][ll] * wr[d];
    outp[(size_t)c * LLEN + l0 + ll] = acc;
  }
}

// ---------------- selective scan: one thread per (b,k,d); 16-state recurrence over L ----------------
// 4 timesteps blocked per iteration: all 39 operand streams are contiguous in l,
// so each becomes one b128 load per 4 steps.
__global__ void k_scan(const float* __restrict__ xs, const float* __restrict__ xdbl,
                       const float* __restrict__ dt_w, const float* __restrict__ dt_b,
                       const float* __restrict__ A_logs, const float* __restrict__ Ds,
                       float* __restrict__ ys) {
  int idx = blockIdx.x * blockDim.x + threadIdx.x;
  if (idx >= BB * KDIR * DI) return;
  int d = idx % DI;
  int k = (idx / DI) % KDIR;
  int b = idx / (DI * KDIR);
  float dtw[RR];
#pragma unroll
  for (int r = 0; r < RR; ++r) dtw[r] = dt_w[(k * DI + d) * RR + r];
  float dtb = dt_b[k * DI + d];
  float Aa[NST];
#pragma unroll
  for (int n = 0; n < NST; ++n) Aa[n] = -__expf(A_logs[(size_t)(k * DI + d) * NST + n]);
  float Dv = Ds[k * DI + d];
  const float* xsp = xs + ((size_t)(b * KDIR + k) * DI + d) * LLEN;
  const float* xd = xdbl + (size_t)(b * KDIR + k) * CPROJ * LLEN;
  float* yp = ys + ((size_t)(b * KDIR + k) * DI + d) * LLEN;
  float hst[NST];
#pragma unroll
  for (int n = 0; n < NST; ++n) hst[n] = 0.f;

  for (int l0 = 0; l0 < LLEN; l0 += 4) {
    float4 u4 = ld4(xsp + l0);
    float4 dd[RR];
#pragma unroll
    for (int r = 0; r < RR; ++r) dd[r] = ld4(xd + (size_t)r * LLEN + l0);
    float4 Bq[NST], Cq[NST];
#pragma unroll
    for (int n = 0; n < NST; ++n) {
      Bq[n] = ld4(xd + (size_t)(RR + n) * LLEN + l0);
      Cq[n] = ld4(xd + (size_t)(RR + NST + n) * LLEN + l0);
    }
    float4 yo;
#pragma unroll
    for (int s = 0; s < 4; ++s) {
      float u = getc(u4, s);
      float dts = dtb;
#pragma unroll
      for (int r = 0; r < RR; ++r) dts += dtw[r] * getc(dd[r], s);
      float delta = (dts > 20.f) ? dts : log1pf(__expf(dts));  // softplus
      float du = delta * u;
      float yv = 0.f;
#pragma unroll
      for (int n = 0; n < NST; ++n) {
        hst[n] = __expf(delta * Aa[n]) * hst[n] + du * getc(Bq[n], s);
        yv += hst[n] * getc(Cq[n], s);
      }
      setc(yo, s, yv + Dv * u);   // fold Ds*xs here (pre-flip, per direction)
    }
    *(float4*)(yp + l0) = yo;
  }
}

// ---------------- combine 4 directions (flips + inverse scatters) -> y(B,L,Di) ----------------
__global__ void k_combine(const float* __restrict__ ys, float* __restrict__ y) {
  int idx = blockIdx.x * blockDim.x + threadIdx.x;
  if (idx >= BB * DI * HH * WW) return;
  int j = idx & (WW - 1);
  int i = (idx >> 6) & (HH - 1);
  int d = (idx >> 12) % DI;
  int b = idx / (DI * HH * WW);
  int l1 = ((j - i + WW) & (WW - 1)) * HH + i;  // diag_inv
  int l2 = ((j + i) & (WW - 1)) * HH + i;       // adiag_inv
  size_t base = ((size_t)(b * KDIR) * DI + d) * LLEN;
  size_t str = (size_t)DI * LLEN;
  float v = ys[base + 0 * str + l1] + ys[base + 2 * str + (LLEN - 1 - l1)]  // yd
          + ys[base + 1 * str + l2] + ys[base + 3 * str + (LLEN - 1 - l2)]; // ya
  y[((size_t)b * LLEN + i * WW + j) * DI + d] = v;
}

// ---------------- out-norm (LN over Di) fused with gating by silu(z); writes f16 for GEMM ----------------
__global__ void k_outnorm_gate(const float* __restrict__ y, const float* __restrict__ xz,
                               const float* __restrict__ g, const float* __restrict__ bta,
                               _Float16* __restrict__ out) {
  int wv = (blockIdx.x * blockDim.x + threadIdx.x) >> 5;
  int lane = threadIdx.x & 31;
  if (wv >= MROWS) return;
  const float* yr = y + (size_t)wv * DI;
  float s = 0.f, s2 = 0.f;
  for (int c = lane; c < DI; c += 32) { float v = yr[c]; s += v; s2 += v * v; }
  for (int o = 16; o; o >>= 1) { s += __shfl_xor(s, o, 32); s2 += __shfl_xor(s2, o, 32); }
  float mu = s / DI;
  float rs = rsqrtf(s2 / DI - mu * mu + 1e-5f);
  const float* z = xz + (size_t)wv * (2 * DI) + DI;
  _Float16* o = out + (size_t)wv * DI;
  for (int c = lane; c < DI; c += 32)
    o[c] = (_Float16)(((yr[c] - mu) * rs * g[c] + bta[c]) * silu_f(z[c]));
}

// ---------------- KAN pre: silu(x) and 8 cubic B-spline bases, written as f16 ----------------
__global__ void k_kan_pre(const float* __restrict__ in, _Float16* __restrict__ S,
                          _Float16* __restrict__ P) {
  int idx = blockIdx.x * blockDim.x + threadIdx.x;
  if (idx >= MROWS * CC) return;
  float v = in[idx];
  S[idx] = (_Float16)silu_f(v);
  float bsp[11];
#pragma unroll
  for (int j = 0; j < 11; ++j) {
    float gj = 0.4f * (j - 3) - 1.0f;
    float gj1 = 0.4f * (j - 2) - 1.0f;
    bsp[j] = (v >= gj && v < gj1) ? 1.0f : 0.0f;
  }
#pragma unroll
  for (int k = 1; k <= 3; ++k) {
    float inv = 1.0f / (0.4f * k);
#pragma unroll
    for (int j = 0; j + k < 11; ++j) {
      float gj = 0.4f * (j - 3) - 1.0f;           // g[j]
      float gjk1 = 0.4f * (j + k - 2) - 1.0f;     // g[j+k+1]
      bsp[j] = (v - gj) * inv * bsp[j] + (gjk1 - v) * inv * bsp[j + 1];
    }
  }
  int m = idx / CC, c = idx % CC;
  v8h pk;
#pragma unroll
  for (int e = 0; e < NB; ++e) pk[e] = (_Float16)bsp[e];
  *(v8h*)(P + (size_t)m * KSPL + c * NB) = pk;    // 16B-aligned
}

// ---------------- depthwise 3x3 conv + (BN-ish scale) + ReLU on (B,L,C) tensor (+resid) ----------------
// c is the fastest thread index: all reads and writes coalesced.
__global__ void k_dw_bn_relu(const float* __restrict__ T, const float* __restrict__ w,
                             const float* __restrict__ bias, const float* __restrict__ gamma,
                             const float* __restrict__ beta, const float* __restrict__ resid,
                             float* __restrict__ out) {
  int idx = blockIdx.x * blockDim.x + threadIdx.x;
  if (idx >= BB * CC * HH * WW) return;
  int c = idx % CC;
  int rest = idx / CC;
  int j = rest & (WW - 1);
  int i = (rest >> 6) & (HH - 1);
  int b = rest >> 12;
  float acc = bias[c];
#pragma unroll
  for (int ki = 0; ki < 3; ++ki) {
#pragma unroll
    for (int kj = 0; kj < 3; ++kj) {
      int ii = i + ki - 1, jj = j + kj - 1;
      if (ii >= 0 && ii < HH && jj >= 0 && jj < WW)
        acc += w[c * 9 + ki * 3 + kj] * T[((size_t)b * LLEN + ii * WW + jj) * CC + c];
    }
  }
  acc = acc * (gamma[c] * rsqrtf(1.0f + 1e-5f)) + beta[c];
  acc = fmaxf(acc, 0.0f);
  size_t o = ((size_t)b * LLEN + i * WW + j) * CC + c;
  if (resid) acc += resid[o];
  out[o] = acc;
}

// ---------------- host-side launch ----------------
static inline int ceil_div(int a, int b) { return (a + b - 1) / b; }

extern "C" void kernel_launch(void* const* d_in, const int* in_sizes, int n_in,
                              void* d_out, int out_size, void* d_ws, size_t ws_size,
                              hipStream_t stream) {
  const float* x         = (const float*)d_in[0];
  const float* norm1_g   = (const float*)d_in[3];
  const float* norm1_b   = (const float*)d_in[4];
  const float* in_proj_w = (const float*)d_in[5];   // (384, 96)
  const float* conv_w    = (const float*)d_in[6];
  const float* conv_b    = (const float*)d_in[7];
  const float* x_proj_w  = (const float*)d_in[8];   // (4, 38, 192)
  const float* dt_w      = (const float*)d_in[9];   // (4, 192, 6)
  const float* dt_b      = (const float*)d_in[10];  // (4, 192)
  const float* A_logs    = (const float*)d_in[11];  // (768, 16)
  const float* Ds        = (const float*)d_in[12];  // (768,)
  const float* onorm_g   = (const float*)d_in[13];
  const float* onorm_b   = (const float*)d_in[14];
  const float* out_proj_w= (const float*)d_in[15];  // (96, 192)
  const float* norm2_g   = (const float*)d_in[16];
  const float* norm2_b   = (const float*)d_in[17];
  const float* fc_base[3]   = { (const float*)d_in[18], (const float*)d_in[20], (const float*)d_in[22] };
  const float* fc_spline[3] = { (const float*)d_in[19], (const float*)d_in[21], (const float*)d_in[23] };
  const float* dw_w[3]    = { (const float*)d_in[24], (const float*)d_in[28], (const float*)d_in[32] };
  const float* dw_b[3]    = { (const float*)d_in[25], (const float*)d_in[29], (const float*)d_in[33] };
  const float* dw_g[3]    = { (const float*)d_in[26], (const float*)d_in[30], (const float*)d_in[34] };
  const float* dw_beta[3] = { (const float*)d_in[27], (const float*)d_in[31], (const float*)d_in[35] };
  float* out = (float*)d_out;

  // workspace layout (f32 region, then f16 arena)
  float* ws = (float*)d_ws;
  size_t oH   = 0;                           // h2 (f32 for KAN pre)        (M*C)
  size_t oXZ  = oH   + (size_t)MROWS * CC;   // xz                          (M*384)
  size_t oIMG = oXZ  + (size_t)MROWS * 2 * DI; // img                       (B*Di*H*W)
  size_t oXS  = oIMG + (size_t)BB * DI * LLEN; // xs (B*4*Di*L) [P16 aliases after scan]
  size_t oXD  = oXS  + (size_t)BB * KDIR * DI * LLEN;    // xdbl (B*4*38*L)
  size_t oYS  = oXD  + (size_t)BB * KDIR * CPROJ * LLEN; // ys   (B*4*Di*L)
  size_t oY   = oYS  + (size_t)BB * KDIR * DI * LLEN;    // y    (M*Di)
  size_t oXM  = oY   + (size_t)MROWS * DI;   // x_mid                       (M*C)
  size_t oT   = oXM  + (size_t)MROWS * CC;   // KAN linear out              (M*C)
  size_t oT2  = oT   + (size_t)MROWS * CC;   // dw out                      (M*C)
  size_t oF16 = oT2  + (size_t)MROWS * CC;   // f16 arena start
  float *hbuf = ws + oH, *xz = ws + oXZ, *img = ws + oIMG, *xs = ws + oXS;
  float *xdbl = ws + oXD, *ysb = ws + oYS, *ybuf = ws + oY, *xmid = ws + oXM;
  float *Tbuf = ws + oT, *T2buf = ws + oT2;

  _Float16* h16  = (_Float16*)(ws + oF16);               // M*C halves
  _Float16* y16  = h16 + (size_t)MROWS * CC;             // M*Di halves
  _Float16* S16  = y16 + (size_t)MROWS * DI;             // M*C halves
  _Float16* wIn  = S16 + (size_t)MROWS * CC;             // weight arena
  _Float16* wOut = wIn + (size_t)(2 * DI) * CC;
  _Float16* wB[3], *wS[3];
  {
    _Float16* p = wOut + (size_t)CC * DI;
    for (int l = 0; l < 3; ++l) { wB[l] = p; p += (size_t)CC * CC;
                                  wS[l] = p; p += (size_t)CC * KSPL; }
  }
  _Float16* P16 = (_Float16*)xs;   // xs dead after k_scan; M*768 halves fit easily

  const int TPB = 256;
  const int WPB = TPB / 32;

  // 0) weights -> f16 (once)
  auto cvtw = [&](const float* src, _Float16* dst, int n) {
    k_cvt_f16<<<ceil_div(n / 4, TPB), TPB, 0, stream>>>(src, dst, n / 4);
  };
  cvtw(in_proj_w, wIn, 2 * DI * CC);
  cvtw(out_proj_w, wOut, CC * DI);
  for (int l = 0; l < 3; ++l) {
    cvtw(fc_base[l], wB[l], CC * CC);
    cvtw(fc_spline[l], wS[l], CC * KSPL);
  }

  // 1) h16 = f16(LN(x))
  k_layernorm<<<ceil_div(MROWS, WPB), TPB, 0, stream>>>(x, norm1_g, norm1_b,
                                                        nullptr, h16, MROWS, CC);
  // 2) xz = h @ in_proj^T    [WMMA f16, 6 N-tiles/wave, K=96]
  {
    int grps = (MROWS / 16) * ((2 * DI) / 96);
    k_wmma_gemm<6, 0, 96><<<ceil_div(grps, WPB), TPB, 0, stream>>>(h16, wIn, nullptr, xz,
                                                                   MROWS, 2 * DI);
  }
  // 3) img = silu(dwconv(xin))
  k_dwconv_silu<<<ceil_div(BB * DI * LLEN, TPB), TPB, 0, stream>>>(xz, conv_w, conv_b, img);
  // 4) xs: 4-direction gathers
  k_build_xs<<<ceil_div(BB * DI * LLEN, TPB), TPB, 0, stream>>>(img, xs);
  // 5) x_dbl = einsum(xs, x_proj_w)  [LDS-staged]
  {
    dim3 grid(LLEN / XT, BB * KDIR);
    k_xproj<<<grid, TPB, 0, stream>>>(xs, x_proj_w, xdbl);
  }
  // 6) selective scan (delta from dts + dt_w/dt_b inline; 4-step blocked b128 streams)
  k_scan<<<ceil_div(BB * KDIR * DI, TPB), TPB, 0, stream>>>(xs, xdbl, dt_w, dt_b, A_logs, Ds, ysb);
  // 7) combine directions -> y(B,L,Di)
  k_combine<<<ceil_div(BB * DI * LLEN, TPB), TPB, 0, stream>>>(ysb, ybuf);
  // 8) out-norm + gate by silu(z) -> f16
  k_outnorm_gate<<<ceil_div(MROWS, WPB), TPB, 0, stream>>>(ybuf, xz, onorm_g, onorm_b, y16);
  // 9) x_mid = x + y @ out_proj^T   [WMMA f16 + residual, 6 N-tiles/wave, K=192]
  {
    int grps = (MROWS / 16) * (CC / 96);
    k_wmma_gemm<6, 1, 192><<<ceil_div(grps, WPB), TPB, 0, stream>>>(y16, wOut, x, xmid,
                                                                    MROWS, CC);
  }
  // 10) h2 = LN(x_mid)  (f32, feeds KAN pre)
  k_layernorm<<<ceil_div(MROWS, WPB), TPB, 0, stream>>>(xmid, norm2_g, norm2_b,
                                                        hbuf, nullptr, MROWS, CC);

  // 11) KAN tower: 3 x (kan_linear [WMMA x2] -> dw+bn+relu); final fused with residual
  const float* cur = hbuf;
  for (int layer = 0; layer < 3; ++layer) {
    k_kan_pre<<<ceil_div(MROWS * CC, TPB), TPB, 0, stream>>>(cur, S16, P16);
    int grps = (MROWS / 16) * (CC / 96);   // groupsN == 1: A streamed once
    k_wmma_gemm<6, 0, 96><<<ceil_div(grps, WPB), TPB, 0, stream>>>(S16, wB[layer], nullptr,
                                                                   Tbuf, MROWS, CC);
    k_wmma_gemm<6, 2, 768><<<ceil_div(grps, WPB), TPB, 0, stream>>>(P16, wS[layer], nullptr,
                                                                    Tbuf, MROWS, CC);
    const float* res = (layer == 2) ? xmid : nullptr;
    float* dst = (layer == 2) ? out : T2buf;
    k_dw_bn_relu<<<ceil_div(BB * CC * LLEN, TPB), TPB, 0, stream>>>(
        Tbuf, dw_w[layer], dw_b[layer], dw_g[layer], dw_beta[layer], res, dst);
    cur = T2buf;
  }
  (void)in_sizes; (void)n_in; (void)out_size; (void)ws_size;
}
